// TransformerLayer_48249662603717
// MI455X (gfx1250) — compile-verified, matrix operand order checked by reading, not run on hardware
//
#include <hip/hip_runtime.h>
#include <hip/hip_bf16.h>
#include <math.h>

// ---------------------------------------------------------------------------
// Types
// ---------------------------------------------------------------------------
typedef __attribute__((ext_vector_type(16))) _Float16 v16h;
typedef __attribute__((ext_vector_type(8)))  _Float16 v8h;
typedef __attribute__((ext_vector_type(8)))  float    v8f;

#define B_   8
#define S_   512
#define D_   1024
#define H_   16
#define HD_  64
#define FF_  4096
#define TOK  (B_ * S_)      // 4096 rows

// ---------------------------------------------------------------------------
// WMMA helper: D = A(16x32 f16) * B(32x16 f16) + C(f32)
// ---------------------------------------------------------------------------
__device__ __forceinline__ v8f wmma32(v16h a, v16h b, v8f c) {
  return __builtin_amdgcn_wmma_f32_16x16x32_f16(false, a, false, b,
                                                (short)0, c, false, false);
}

// Load one A/B operand fragment (v16h) for a 32-wide K chunk.
// p must already point at element [row][k0 + ko8] where
//   row = lane & 15, ko8 = (lane >> 4) * 8.
// Fragment = 8 contiguous f16 at p, plus 8 contiguous f16 at p+16.
__device__ __forceinline__ v16h frag_ld(const _Float16* p) {
  v8h lo = *(const v8h*)(p);
  v8h hi = *(const v8h*)(p + 16);
  v16h r;
#pragma unroll
  for (int i = 0; i < 8; ++i) { r[i] = lo[i]; r[i + 8] = hi[i]; }
  return r;
}

// ---------------------------------------------------------------------------
// Weight packing: f32 [K][N] row-major  ->  f16 [N][K] ("B operand" layout)
// ---------------------------------------------------------------------------
__global__ void pack_T_kernel(const float* __restrict__ in,
                              _Float16* __restrict__ out,
                              int K, int N, size_t total) {
  size_t o = (size_t)blockIdx.x * blockDim.x + threadIdx.x;
  if (o >= total) return;
  size_t n = o / (size_t)K;
  size_t k = o - n * (size_t)K;
  out[o] = (_Float16)in[k * (size_t)N + n];
}

// Pack per-head Wq/Wk/Wv (H,D,HD) into fused [3*D][D] f16, [n][k] layout,
// column n = qkv*D + h*HD + e.
__global__ void pack_qkv_kernel(const float* __restrict__ Wq,
                                const float* __restrict__ Wk,
                                const float* __restrict__ Wv,
                                _Float16* __restrict__ out) {
  size_t o = (size_t)blockIdx.x * blockDim.x + threadIdx.x;
  if (o >= (size_t)3 * D_ * D_) return;
  int k  = (int)(o & (D_ - 1));
  int n  = (int)(o >> 10);
  int qkv = n >> 10;
  int wn  = n & (D_ - 1);
  int h   = wn >> 6;
  int e   = wn & (HD_ - 1);
  const float* W = (qkv == 0) ? Wq : (qkv == 1) ? Wk : Wv;
  out[o] = (_Float16)W[((size_t)h * D_ + k) * HD_ + e];
}

// ---------------------------------------------------------------------------
// LayerNorm: f32 row [D] -> f16 row [D]  (biased variance, eps=1e-5)
// one row per block, 256 threads, 4 elems/thread
// ---------------------------------------------------------------------------
__global__ void ln_kernel(const float* __restrict__ x,
                          const float* __restrict__ g,
                          const float* __restrict__ b,
                          _Float16* __restrict__ out) {
  int row = blockIdx.x;
  const float4 v = ((const float4*)(x + (size_t)row * D_))[threadIdx.x];
  float s  = v.x + v.y + v.z + v.w;
  float s2 = v.x * v.x + v.y * v.y + v.z * v.z + v.w * v.w;
#pragma unroll
  for (int o = 16; o; o >>= 1) {
    s  += __shfl_down(s,  o, 32);
    s2 += __shfl_down(s2, o, 32);
  }
  __shared__ float sh[16];
  int wid = threadIdx.x >> 5;
  if ((threadIdx.x & 31) == 0) { sh[wid * 2] = s; sh[wid * 2 + 1] = s2; }
  __syncthreads();
  float ts = 0.f, ts2 = 0.f;
#pragma unroll
  for (int i = 0; i < 8; ++i) { ts += sh[i * 2]; ts2 += sh[i * 2 + 1]; }
  float mean = ts * (1.f / D_);
  float var  = ts2 * (1.f / D_) - mean * mean;
  float rstd = rsqrtf(var + 1e-5f);
  int c0 = threadIdx.x * 4;
  _Float16* orow = out + (size_t)row * D_;
  float xv[4] = {v.x, v.y, v.z, v.w};
#pragma unroll
  for (int i = 0; i < 4; ++i)
    orow[c0 + i] = (_Float16)((xv[i] - mean) * rstd * g[c0 + i] + b[c0 + i]);
}

// ---------------------------------------------------------------------------
// GEMM epilogue for one 16x16 C tile (explicit, no pointer arrays)
// ---------------------------------------------------------------------------
template <int MODE>
__device__ __forceinline__ void epilogue_tile(
    v8f c, int tile_m, int tile_n, int lane, int N,
    const float* __restrict__ bias, const float* __restrict__ res,
    float* __restrict__ outf, _Float16* __restrict__ outh,
    _Float16* __restrict__ qo, _Float16* __restrict__ ko,
    _Float16* __restrict__ vo) {
  const int hi8 = (lane >> 4) << 3;
  const int n   = tile_n + (lane & 15);
#pragma unroll
  for (int j = 0; j < 8; ++j) {
    int m = tile_m + j + hi8;
    float cv = c[j];
    if (MODE == 0) {
      int qkv = n >> 10, wn = n & (D_ - 1);
      int h = wn >> 6, e = wn & (HD_ - 1);
      int bb = m >> 9, ss = m & (S_ - 1);
      _Float16 hv = (_Float16)cv;
      size_t rmaj = (((size_t)(bb * H_ + h)) * S_ + ss) * HD_ + e; // Q,K
      size_t cmaj = (((size_t)(bb * H_ + h)) * HD_ + e) * S_ + ss; // V^T
      if (qkv == 0)      qo[rmaj] = hv;
      else if (qkv == 1) ko[rmaj] = hv;
      else               vo[cmaj] = hv;
    } else if (MODE == 1) {
      size_t o = (size_t)m * N + n;
      outf[o] = cv + bias[n] + res[o];
    } else { // MODE 2: bias + exact GELU -> f16
      float xv = cv + bias[n];
      float gg = 0.5f * xv * (1.f + erff(xv * 0.70710678118654752f));
      outh[(size_t)m * N + n] = (_Float16)gg;
    }
  }
}

// ---------------------------------------------------------------------------
// Generic WMMA GEMM:  C[M,N] = A[M,K] (f16, row-major) * Bp[N,K] (f16)
// Block = 256 thr = 8 waves; wave = 32x32 tile (2x2 WMMA tiles);
// block tile = 64 (M) x 128 (N). Grid = (M/64, N/128).
// Inner K loop is software-pipelined: fragments for k+32 are loaded before
// the WMMAs for k are issued, so s_wait_loadcnt overlaps 4 WMMAs of work.
// ---------------------------------------------------------------------------
template <int MODE>
__global__ void __launch_bounds__(256, 1)
gemm_kernel(const _Float16* __restrict__ A,
            const _Float16* __restrict__ Bp,
            int K, int N,
            const float* __restrict__ bias,
            const float* __restrict__ res,
            float* __restrict__ outf,
            _Float16* __restrict__ outh,
            _Float16* __restrict__ qo,
            _Float16* __restrict__ ko,
            _Float16* __restrict__ vo) {
  const int lane = threadIdx.x & 31;
  const int wid  = threadIdx.x >> 5;
  const int tm   = blockIdx.x * 64  + (wid & 1) * 32;
  const int tn   = blockIdx.y * 128 + (wid >> 1) * 32;
  const int row  = lane & 15;
  const int ko8  = (lane >> 4) << 3;

  v8f acc00 = {}, acc01 = {}, acc10 = {}, acc11 = {};

  const _Float16* ap = A  + (size_t)(tm + row) * K + ko8;
  const _Float16* bp = Bp + (size_t)(tn + row) * K + ko8;
  const size_t a16 = (size_t)16 * K;

  // prologue: fragments for kk = 0
  v16h a0 = frag_ld(ap);
  v16h a1 = frag_ld(ap + a16);
  v16h b0 = frag_ld(bp);
  v16h b1 = frag_ld(bp + a16);

  for (int kk = 0; kk < K; kk += 32) {
    const int kn = kk + 32;
    v16h na0, na1, nb0, nb1;
    if (kn < K) {
      __builtin_prefetch(ap + kn + 32, 0, 3);
      __builtin_prefetch(bp + kn + 32, 0, 3);
      na0 = frag_ld(ap + kn);
      na1 = frag_ld(ap + kn + a16);
      nb0 = frag_ld(bp + kn);
      nb1 = frag_ld(bp + kn + a16);
    }
    acc00 = wmma32(a0, b0, acc00);
    acc01 = wmma32(a0, b1, acc01);
    acc10 = wmma32(a1, b0, acc10);
    acc11 = wmma32(a1, b1, acc11);
    if (kn < K) { a0 = na0; a1 = na1; b0 = nb0; b1 = nb1; }
  }

  epilogue_tile<MODE>(acc00, tm,      tn,      lane, N, bias, res, outf, outh, qo, ko, vo);
  epilogue_tile<MODE>(acc01, tm,      tn + 16, lane, N, bias, res, outf, outh, qo, ko, vo);
  epilogue_tile<MODE>(acc10, tm + 16, tn,      lane, N, bias, res, outf, outh, qo, ko, vo);
  epilogue_tile<MODE>(acc11, tm + 16, tn + 16, lane, N, bias, res, outf, outh, qo, ko, vo);
}

// ---------------------------------------------------------------------------
// Flash attention (causal). Q,K: [B,H,S,HD] f16. Vt: [B,H,HD,S] f16.
// ctx out: [B,S,H*HD] f16 (row-major, ready as A for the Wo GEMM).
// Block = 256 thr = 8 waves; wave owns a 16-row Q tile; grid = (B*H, S/128).
// Score sub-tiles computed sequentially so K fragments die immediately
// (shrinks peak VGPR liveness; with launch_bounds there should be no spill).
// ---------------------------------------------------------------------------
__global__ void __launch_bounds__(256, 1)
attn_kernel(const _Float16* __restrict__ Q,
            const _Float16* __restrict__ Kf,
            const _Float16* __restrict__ Vt,
            _Float16* __restrict__ ctx) {
  __shared__ _Float16 lds_p[8][16 * 40]; // wave-private 16x32 P tile, pad 40

  const int lane = threadIdx.x & 31;
  const int wid  = threadIdx.x >> 5;
  const int bh   = blockIdx.x;          // b*H + h
  const int b    = bh >> 4;
  const int h    = bh & (H_ - 1);
  const int q0   = blockIdx.y * 128 + wid * 16;
  const int row  = lane & 15;
  const int ko8  = (lane >> 4) << 3;
  const int hi8  = ko8;

  const _Float16* qb = Q  + (size_t)bh * S_ * HD_;
  const _Float16* kb = Kf + (size_t)bh * S_ * HD_;
  const _Float16* vb = Vt + (size_t)bh * HD_ * S_;

  // Q A-operand fragments (held for the whole loop)
  const _Float16* qp = qb + (size_t)(q0 + row) * HD_ + ko8;
  const v16h qa0 = frag_ld(qp);        // e = 0..31
  const v16h qa1 = frag_ld(qp + 32);   // e = 32..63

  float m_r[8], l_r[8], corr_r[8];
  v8f acc0 = {}, acc1 = {}, acc2 = {}, acc3 = {};
#pragma unroll
  for (int j = 0; j < 8; ++j) { m_r[j] = -1e30f; l_r[j] = 0.f; }

  for (int kblk = 0; kblk <= q0 + 15; kblk += 32) {
    // ---- scores: S(16x32) = Q(16x64) * K^T, two 16x16 tiles in sequence ----
    const _Float16* kp = kb + (size_t)(kblk + row) * HD_ + ko8;
    v8f s0 = {};
    {
      v16h k00 = frag_ld(kp);               // keys kblk..+15, e 0..31
      v16h k01 = frag_ld(kp + 32);          //               e 32..63
      s0 = wmma32(qa0, k00, s0);
      s0 = wmma32(qa1, k01, s0);
    }
    v8f s1 = {};
    {
      v16h k10 = frag_ld(kp + 16 * HD_);    // keys +16..+31
      v16h k11 = frag_ld(kp + 16 * HD_ + 32);
      s1 = wmma32(qa0, k10, s1);
      s1 = wmma32(qa1, k11, s1);
    }

    // ---- masked online softmax (rows live in 16-lane halves) ----
#pragma unroll
    for (int j = 0; j < 8; ++j) {
      int qg  = q0 + j + hi8;
      int kg0 = kblk + (lane & 15);
      int kg1 = kg0 + 16;
      float v0 = s0[j] * 0.125f; if (kg0 > qg) v0 = -1e10f;
      float v1 = s1[j] * 0.125f; if (kg1 > qg) v1 = -1e10f;
      float mx = fmaxf(v0, v1);
#pragma unroll
      for (int o = 8; o; o >>= 1) mx = fmaxf(mx, __shfl_xor(mx, o, 32));
      float mn = fmaxf(m_r[j], mx);
      float p0 = __expf(v0 - mn);
      float p1 = __expf(v1 - mn);
      float sm = p0 + p1;
#pragma unroll
      for (int o = 8; o; o >>= 1) sm += __shfl_xor(sm, o, 32);
      corr_r[j] = __expf(m_r[j] - mn);
      l_r[j]    = l_r[j] * corr_r[j] + sm;
      m_r[j]    = mn;
      int lr = j + hi8;
      lds_p[wid][lr * 40 + (lane & 15)]      = (_Float16)p0;
      lds_p[wid][lr * 40 + 16 + (lane & 15)] = (_Float16)p1;
    }
#pragma unroll
    for (int j = 0; j < 8; ++j) {
      acc0[j] *= corr_r[j]; acc1[j] *= corr_r[j];
      acc2[j] *= corr_r[j]; acc3[j] *= corr_r[j];
    }
    asm volatile("s_wait_dscnt 0" ::: "memory");

    // ---- ctx += P(16x32) * V(32x64) ----
    v16h pa = frag_ld(&lds_p[wid][(lane & 15) * 40 + ko8]);
    const _Float16* vp = vb + (size_t)(lane & 15) * S_ + kblk + ko8;
    acc0 = wmma32(pa, frag_ld(vp),                   acc0);
    acc1 = wmma32(pa, frag_ld(vp + (size_t)16 * S_), acc1);
    acc2 = wmma32(pa, frag_ld(vp + (size_t)32 * S_), acc2);
    acc3 = wmma32(pa, frag_ld(vp + (size_t)48 * S_), acc3);
  }

  // ---- normalize + write ctx [B,S,D] f16 (explicit, no pointer arrays) ----
  float inv[8];
#pragma unroll
  for (int j = 0; j < 8; ++j) inv[j] = 1.f / l_r[j];
  const size_t rowbase = (size_t)b * S_ + q0;
#pragma unroll
  for (int j = 0; j < 8; ++j) {
    size_t ro = (rowbase + j + hi8) * D_ + h * HD_ + (lane & 15);
    ctx[ro]      = (_Float16)(acc0[j] * inv[j]);
    ctx[ro + 16] = (_Float16)(acc1[j] * inv[j]);
    ctx[ro + 32] = (_Float16)(acc2[j] * inv[j]);
    ctx[ro + 48] = (_Float16)(acc3[j] * inv[j]);
  }
}

// ---------------------------------------------------------------------------
// Host-side launch
// ---------------------------------------------------------------------------
extern "C" void kernel_launch(void* const* d_in, const int* in_sizes, int n_in,
                              void* d_out, int out_size, void* d_ws, size_t ws_size,
                              hipStream_t stream) {
  const float* x   = (const float*)d_in[0];
  const float* Wq  = (const float*)d_in[1];
  const float* Wk  = (const float*)d_in[2];
  const float* Wv  = (const float*)d_in[3];
  const float* Wo  = (const float*)d_in[4];
  const float* bo  = (const float*)d_in[5];
  const float* g1  = (const float*)d_in[6];
  const float* b1  = (const float*)d_in[7];
  const float* g2  = (const float*)d_in[8];
  const float* b2  = (const float*)d_in[9];
  const float* W1  = (const float*)d_in[10];
  const float* bf1 = (const float*)d_in[11];
  const float* W2  = (const float*)d_in[12];
  const float* bf2 = (const float*)d_in[13];
  float* out = (float*)d_out;

  char* ws = (char*)d_ws;
  size_t off = 0;
  auto alloc = [&](size_t bytes) -> char* {
    char* p = ws + off;
    off += (bytes + 255) & ~(size_t)255;
    return p;
  };
  _Float16* hln1  = (_Float16*)alloc((size_t)TOK * D_ * 2);
  _Float16* Wqkv  = (_Float16*)alloc((size_t)3 * D_ * D_ * 2);
  _Float16* Qb    = (_Float16*)alloc((size_t)TOK * D_ * 2);
  _Float16* Kb    = (_Float16*)alloc((size_t)TOK * D_ * 2);
  _Float16* Vt    = (_Float16*)alloc((size_t)TOK * D_ * 2);
  _Float16* ctx   = (_Float16*)alloc((size_t)TOK * D_ * 2);
  _Float16* Wop   = (_Float16*)alloc((size_t)D_ * D_ * 2);
  float*    cache = (float*)   alloc((size_t)TOK * D_ * 4);
  _Float16* h2    = (_Float16*)alloc((size_t)TOK * D_ * 2);
  _Float16* W1p   = (_Float16*)alloc((size_t)D_ * FF_ * 2);
  _Float16* ff1   = (_Float16*)alloc((size_t)TOK * FF_ * 2);
  _Float16* W2p   = (_Float16*)alloc((size_t)FF_ * D_ * 2);
  (void)ws_size; (void)in_sizes; (void)n_in; (void)out_size;

  // 1) pack weights to f16 [N][K]
  {
    size_t t = (size_t)3 * D_ * D_;
    pack_qkv_kernel<<<dim3((unsigned)((t + 255) / 256)), 256, 0, stream>>>(Wq, Wk, Wv, Wqkv);
  }
  {
    size_t t = (size_t)D_ * D_;
    pack_T_kernel<<<dim3((unsigned)((t + 255) / 256)), 256, 0, stream>>>(Wo, Wop, D_, D_, t);
  }
  {
    size_t t = (size_t)D_ * FF_;
    pack_T_kernel<<<dim3((unsigned)((t + 255) / 256)), 256, 0, stream>>>(W1, W1p, D_, FF_, t);
    pack_T_kernel<<<dim3((unsigned)((t + 255) / 256)), 256, 0, stream>>>(W2, W2p, FF_, D_, t);
  }

  // 2) LN1
  ln_kernel<<<dim3(TOK), 256, 0, stream>>>(x, g1, b1, hln1);

  // 3) fused QKV projection (scatter; V transposed)
  gemm_kernel<0><<<dim3(TOK / 64, (3 * D_) / 128), 256, 0, stream>>>(
      hln1, Wqkv, D_, 3 * D_, nullptr, nullptr, nullptr, nullptr, Qb, Kb, Vt);

  // 4) causal flash attention
  attn_kernel<<<dim3(B_ * H_, S_ / 128), 256, 0, stream>>>(Qb, Kb, Vt, ctx);

  // 5) output projection + bias + residual -> cache (f32)
  gemm_kernel<1><<<dim3(TOK / 64, D_ / 128), 256, 0, stream>>>(
      ctx, Wop, D_, D_, bo, x, cache, nullptr, nullptr, nullptr, nullptr);

  // 6) LN2
  ln_kernel<<<dim3(TOK), 256, 0, stream>>>(cache, g2, b2, h2);

  // 7) FFN up + GELU -> f16
  gemm_kernel<2><<<dim3(TOK / 64, FF_ / 128), 256, 0, stream>>>(
      h2, W1p, D_, FF_, bf1, nullptr, nullptr, ff1, nullptr, nullptr, nullptr);

  // 8) FFN down + bias + residual(cache) -> d_out (f32)
  gemm_kernel<1><<<dim3(TOK / 64, D_ / 128), 256, 0, stream>>>(
      ff1, W2p, FF_, D_, bf2, cache, out, nullptr, nullptr, nullptr, nullptr);
}
